// GCN_13305808683148
// MI455X (gfx1250) — matrix operand order repaired and checked by reading, not compiled, not run on hardware
//
#include <hip/hip_runtime.h>

// ---------------------------------------------------------------------------
// 2-layer GCN for MI455X (gfx1250, wave32, WMMA).
//
// Roofline: adj (400MB f32) must be streamed twice (the two adj-GEMMs cannot
// fuse since S2 depends on all of x1) -> ~34us HBM floor @23.3TB/s. Total
// matrix math ~64 GFLOP; v_wmma_f32_16x16x32_f16 (f16 in, f32 accumulate)
// keeps compute far below that floor (f32 WMMA 16x16x4 would not). adj stays
// f32 in HBM, converted to f16 in registers on the way to LDS. S1/x1/S2 are
// f16 in workspace (<=5.1MB each, L2-resident).
//
// Structure: 128x64 block tile, K-step 32, 8 waves each owning a 32x32
// output macro-tile (4 WMMA / 8 ds_load_b128 per wave per step). Software-
// pipelined global loads (raw staging regs, convert at LDS-store time),
// ping-pong LDS with ONE barrier per K-step, uniform-branch tail handling,
// global_prefetch_b8 two tiles ahead on the adj stream.
// ---------------------------------------------------------------------------

typedef __attribute__((ext_vector_type(16))) _Float16 v16h;
typedef __attribute__((ext_vector_type(8)))  _Float16 v8h;
typedef __attribute__((ext_vector_type(4)))  _Float16 v4h;
typedef __attribute__((ext_vector_type(2)))  _Float16 v2h;
typedef __attribute__((ext_vector_type(8)))  float    v8f;
typedef __attribute__((ext_vector_type(4)))  float    vf4;

#define BM 128
#define BN 64
#define BK 32
#define LDSTR 40   // padded LDS K-stride in halfs: 80B rows, b128-aligned, conflict-spread

// ---- raw staging (convert later, at LDS-store time) ----
struct AStageF { vf4 d0, d1, d2, d3; };   // 16 f32
struct AStageH { v8h d0, d1; };           // 16 f16
struct BStageF { vf4 r0, r1; };           // 2 rows x 4 f32
struct BStageH { v4h r0, r1; };           // 2 rows x 4 f16

template<typename T> struct StagePick;
template<> struct StagePick<float>    { using A = AStageF; using B = BStageF; };
template<> struct StagePick<_Float16> { using A = AStageH; using B = BStageH; };

// A: 16 contiguous K elems of one row. K % 16 == 0 for all passes -> whole-chunk guard.
template<bool G>
__device__ __forceinline__ void loadA(const float* p, int gk, int K, AStageF& s) {
  if constexpr (G) { if (gk >= K) { s.d0 = {}; s.d1 = {}; s.d2 = {}; s.d3 = {}; return; } }
  s.d0 = *(const vf4*)(p + gk);
  s.d1 = *(const vf4*)(p + gk + 4);
  s.d2 = *(const vf4*)(p + gk + 8);
  s.d3 = *(const vf4*)(p + gk + 12);
}
template<bool G>
__device__ __forceinline__ void loadA(const _Float16* p, int gk, int K, AStageH& s) {
  if constexpr (G) { if (gk >= K) { s.d0 = {}; s.d1 = {}; return; } }
  s.d0 = *(const v8h*)(p + gk);
  s.d1 = *(const v8h*)(p + gk + 8);
}
__device__ __forceinline__ void storeA(const AStageF& s, _Float16* dst) {
  v8h h0, h1;
  h0[0]=(_Float16)s.d0[0]; h0[1]=(_Float16)s.d0[1]; h0[2]=(_Float16)s.d0[2]; h0[3]=(_Float16)s.d0[3];
  h0[4]=(_Float16)s.d1[0]; h0[5]=(_Float16)s.d1[1]; h0[6]=(_Float16)s.d1[2]; h0[7]=(_Float16)s.d1[3];
  h1[0]=(_Float16)s.d2[0]; h1[1]=(_Float16)s.d2[1]; h1[2]=(_Float16)s.d2[2]; h1[3]=(_Float16)s.d2[3];
  h1[4]=(_Float16)s.d3[0]; h1[5]=(_Float16)s.d3[1]; h1[6]=(_Float16)s.d3[2]; h1[7]=(_Float16)s.d3[3];
  *(v8h*)dst = h0;
  *(v8h*)(dst + 8) = h1;
}
__device__ __forceinline__ void storeA(const AStageH& s, _Float16* dst) {
  *(v8h*)dst = s.d0;
  *(v8h*)(dst + 8) = s.d1;
}

// B: rows gk, gk+1 (K even, gk even -> pair-uniform guard), 4 contiguous cols.
template<bool G>
__device__ __forceinline__ void loadB(const float* p, int gk, int Nn, int K, BStageF& s) {
  if constexpr (G) { if (gk >= K) { s.r0 = {}; s.r1 = {}; return; } }
  s.r0 = *(const vf4*)(p + (size_t)gk * Nn);
  s.r1 = *(const vf4*)(p + (size_t)(gk + 1) * Nn);
}
template<bool G>
__device__ __forceinline__ void loadB(const _Float16* p, int gk, int Nn, int K, BStageH& s) {
  if constexpr (G) { if (gk >= K) { s.r0 = {}; s.r1 = {}; return; } }
  s.r0 = *(const v4h*)(p + (size_t)gk * Nn);
  s.r1 = *(const v4h*)(p + (size_t)(gk + 1) * Nn);
}
template<typename ST>
__device__ __forceinline__ void storeB(const ST& s, _Float16* bsBase, int nq, int kp) {
  #pragma unroll
  for (int i = 0; i < 4; ++i) {     // 4x packed b32 LDS stores (k, k+1 adjacent)
    v2h t;
    t[0] = (_Float16)s.r0[i];
    t[1] = (_Float16)s.r1[i];
    *(v2h*)(bsBase + (nq + i) * LDSTR + 2 * kp) = t;
  }
}

// fragment: two b128 LDS loads -> v16h (per ISA per-lane K layouts)
__device__ __forceinline__ v16h frag16(const _Float16* p, int hiOff) {
  v8h lo = *(const v8h*)p;
  v8h hi = *(const v8h*)(p + hiOff);
  return __builtin_shufflevector(lo, hi, 0,1,2,3,4,5,6,7,8,9,10,11,12,13,14,15);
}
__device__ __forceinline__ v8f wmma(v16h a, v16h b, v8f c) {
  return __builtin_amdgcn_wmma_f32_16x16x32_f16(false, a, false, b, (short)0, c, false, false);
}

// EPI: 0 = store f16 (S1/S2)
//      1 = +bias, relu, store f32 (d_out) AND f16 (ws)   [x1]
//      3 = +bias, row-wise log_softmax over 64 cols, f32 [final]
template<int EPI, typename AT, typename BT>
__launch_bounds__(256)
__global__ void gcn_gemm_wmma(const AT* __restrict__ A, const BT* __restrict__ B,
                              const float* __restrict__ bias,
                              float* __restrict__ outF, _Float16* __restrict__ outH,
                              int M, int Nn, int K)
{
  __shared__ _Float16 As[2][BM * LDSTR];               // [row][k], ping-pong
  __shared__ _Float16 Bs[2][BN * LDSTR];               // [col][k], ping-pong
  __shared__ float    Ct[(EPI == 3) ? (BM * 65) : 1];  // softmax staging

  const int tid  = threadIdx.x;
  const int lane = tid & 31;
  const int wave = tid >> 5;      // 8 waves
  const int wm   = wave >> 1;     // 0..3 : 32-row macro-slab
  const int wn   = wave & 1;      // 0..1 : 32-col macro-slab
  const int fr   = lane & 15;
  const int kbA  = (lane < 16) ? 0 : 8;   // ISA 16-bit A frag: halves {kb..kb+7, kb+16..kb+23}
  const int kbB  = (lane < 16) ? 0 : 8;   // B frag same K layout as A: halves {kb..kb+7, kb+16..kb+23}

  const int rowBase = blockIdx.y * BM;
  const int nBase   = blockIdx.x * BN;

  // cooperative-load mapping
  const int aRow = tid >> 1;            // 0..127 : one row per thread
  const int aSeg = (tid & 1) * 16;      // 0 or 16
  const int bKp  = tid >> 4;            // 0..15  : k-pair (k = 2*bKp, 2*bKp+1)
  const int bNq  = (tid & 15) * 4;      // 0..60  : n-quad

  const int gARow = rowBase + aRow;
  // clamp invalid rows to row 0: loads always legal, results never stored
  const AT* pA = A + (size_t)((gARow < M) ? gARow : 0) * K;
  const BT* pB = B + (size_t)nBase + bNq;

  typename StagePick<AT>::A sa;
  typename StagePick<BT>::B sb;
  loadA<false>(pA, aSeg, K, sa);            // K >= BK always
  loadB<false>(pB, 2 * bKp, Nn, K, sb);

  v8f c00 = {}, c01 = {}, c10 = {}, c11 = {};

  int p = 0;
  for (int k0 = 0; k0 < K; k0 += BK) {
    // convert + commit staged tile (global-load waits land here, under prev WMMAs)
    storeA(sa, &As[p][aRow * LDSTR + aSeg]);
    storeB(sb, &Bs[p][0], bNq, bKp);
    __syncthreads();   // single barrier: next store goes to the other buffer

    // issue next tile's global loads (uniform scalar branches only)
    const int kn = k0 + BK;
    if (kn < K) {
      if (kn + BK <= K) {
        loadA<false>(pA, kn + aSeg, K, sa);
        loadB<false>(pB, kn + 2 * bKp, Nn, K, sb);
        __builtin_prefetch(pA + kn + aSeg + BK, 0, 1);   // 2 tiles ahead on adj stream
      } else {
        loadA<true>(pA, kn + aSeg, K, sa);
        loadB<true>(pB, kn + 2 * bKp, Nn, K, sb);
      }
    }

    // fragments + 4 WMMAs (branch-free)
    const _Float16* asb = As[p];
    const _Float16* bsb = Bs[p];
    v16h a0 = frag16(&asb[(wm * 32 + fr)      * LDSTR + kbA], 16);
    v16h a1 = frag16(&asb[(wm * 32 + 16 + fr) * LDSTR + kbA], 16);
    v16h b0 = frag16(&bsb[(wn * 32 + fr)      * LDSTR + kbB], 16);
    v16h b1 = frag16(&bsb[(wn * 32 + 16 + fr) * LDSTR + kbB], 16);

    c00 = wmma(a0, b0, c00);
    c01 = wmma(a0, b1, c01);
    c10 = wmma(a1, b0, c10);
    c11 = wmma(a1, b1, c11);
    p ^= 1;
  }

  // ---- epilogue ----
  const int rAdd = (lane < 16) ? 0 : 8;  // C layout: VGPR r -> row r / 8+r
  auto epi = [&](const v8f& c, int hr, int sc) {
    const int lcol = wn * 32 + sc * 16 + fr;
    const int gc   = nBase + lcol;
    #pragma unroll
    for (int r = 0; r < 8; ++r) {
      const int lrow = wm * 32 + hr * 16 + rAdd + r;
      const int grow = rowBase + lrow;
      float v = c[r];
      if constexpr (EPI == 0) {
        if (grow < M) outH[(size_t)grow * Nn + gc] = (_Float16)v;
      } else if constexpr (EPI == 1) {
        v = fmaxf(v + bias[gc], 0.0f);
        if (grow < M) {
          outF[(size_t)grow * Nn + gc] = v;
          outH[(size_t)grow * Nn + gc] = (_Float16)v;
        }
      } else {  // EPI == 3
        Ct[lrow * 65 + lcol] = v + bias[gc];
      }
    }
  };
  epi(c00, 0, 0); epi(c01, 0, 1); epi(c10, 1, 0); epi(c11, 1, 1);

  if constexpr (EPI == 3) {
    __syncthreads();
    if (tid < BM) {
      const int grow = rowBase + tid;
      if (grow < M) {
        const float* row = &Ct[tid * 65];
        float m = row[0];
        #pragma unroll 8
        for (int j = 1; j < 64; ++j) m = fmaxf(m, row[j]);
        float ssum = 0.0f;
        #pragma unroll 8
        for (int j = 0; j < 64; ++j) ssum += __expf(row[j] - m);
        const float lse = m + __logf(ssum);
        #pragma unroll 8
        for (int j = 0; j < 64; ++j)
          outF[(size_t)grow * 64 + j] = row[j] - lse;
      }
    }
  }
}

extern "C" void kernel_launch(void* const* d_in, const int* in_sizes, int n_in,
                              void* d_out, int out_size, void* d_ws, size_t ws_size,
                              hipStream_t stream) {
  (void)in_sizes; (void)n_in; (void)out_size; (void)ws_size;

  const float* feature = (const float*)d_in[0];   // [10000,512]
  const float* adj     = (const float*)d_in[1];   // [10000,10000]
  const float* W1      = (const float*)d_in[2];   // [512,256]
  const float* b1      = (const float*)d_in[3];   // [256]
  const float* W2      = (const float*)d_in[4];   // [256,64]
  const float* b2      = (const float*)d_in[5];   // [64]

  const int N = 10000, FIN = 512, HID = 256, NCLS = 64;

  float* x1_out = (float*)d_out;                       // [10000,256] f32
  float* ls_out = x1_out + (size_t)N * HID;            // [10000,64]  f32

  // workspace: f16 intermediates (~11.5 MB total, L2-resident)
  _Float16* S1h = (_Float16*)d_ws;                                  // [10000,256]
  _Float16* x1h = (_Float16*)((char*)d_ws + (size_t)N * HID * 2);   // [10000,256]
  _Float16* S2h = (_Float16*)((char*)d_ws + (size_t)N * HID * 4);   // [10000,64]

  const dim3 blk(256);
  const int rb = (N + BM - 1) / BM;   // 79 row blocks (10000 = 78*128 + 16, tail guarded)

  // S1 = feature @ W1              (f32 x f32 -> f16 ws)
  gcn_gemm_wmma<0, float, float><<<dim3(HID / BN, rb), blk, 0, stream>>>(
      feature, W1, nullptr, nullptr, S1h, N, HID, FIN);
  // x1 = relu(adj @ S1 + b1)       (f32 x f16 -> f32 d_out + f16 ws)
  gcn_gemm_wmma<1, float, _Float16><<<dim3(HID / BN, rb), blk, 0, stream>>>(
      adj, S1h, b1, x1_out, x1h, N, HID, N);
  // S2 = x1 @ W2                   (f16 x f32 -> f16 ws)
  gcn_gemm_wmma<0, _Float16, float><<<dim3(NCLS / BN, rb), blk, 0, stream>>>(
      x1h, W2, nullptr, nullptr, S2h, N, NCLS, HID);
  // out2 = log_softmax(adj @ S2 + b2)
  gcn_gemm_wmma<3, float, _Float16><<<dim3(NCLS / BN, rb), blk, 0, stream>>>(
      adj, S2h, b2, ls_out, nullptr, N, NCLS, N);
}
